// IterativeVoxelModel_84507776516328
// MI455X (gfx1250) — compile-verified, matrix operand order checked by reading, not run on hardware
//
#include <hip/hip_runtime.h>
#include <hip/hip_bf16.h>

typedef __attribute__((ext_vector_type(16))) _Float16 v16h;
typedef __attribute__((ext_vector_type(8)))  _Float16 v8h;
typedef __attribute__((ext_vector_type(8)))  float    v8f;

#define WSZ 3
#define VOX 27
#define NB  256          // batch
#define NTOK (NB * VOX)  // 6912 tokens
#define DM  256
#define FF  1024
#define NH  8
#define HD  32
#define NLAYERS 3

union FragU { v16h f; v8h h[2]; };

// ---------------------------------------------------------------------------
// Weight pack: f32 [K][Nt] row-major -> f16 [Nt][K] (transposed, N-major)
// ---------------------------------------------------------------------------
__global__ void pack_wT_kernel(const float* __restrict__ src, _Float16* __restrict__ dst,
                               int K, int Nt) {
  int gid = blockIdx.x * blockDim.x + threadIdx.x;
  if (gid >= K * Nt) return;
  int n = gid / K;
  int k = gid - n * K;
  dst[n * K + k] = (_Float16)src[k * Nt + n];
}

// ---------------------------------------------------------------------------
// Embed: t[n][c] = patch[n] * w_in[c] + b_in[c] + pos[cell][c]
// ---------------------------------------------------------------------------
__global__ void embed_kernel(const float* __restrict__ patch,
                             const float* __restrict__ w_in,
                             const float* __restrict__ b_in,
                             const float* __restrict__ pos,
                             float* __restrict__ t32, _Float16* __restrict__ t16) {
  int n = blockIdx.x;        // token
  int c = threadIdx.x;       // channel
  int cell = n % VOX;
  float val = patch[n] * w_in[c] + b_in[c] + pos[cell * DM + c];
  t32[n * DM + c] = val;
  t16[n * DM + c] = (_Float16)val;
}

// ---------------------------------------------------------------------------
// LayerNorm over 256 channels; one token per 256-thread block; LDS reduction
// ---------------------------------------------------------------------------
__global__ void layernorm_kernel(const float* __restrict__ t32,
                                 const float* __restrict__ g,
                                 const float* __restrict__ b,
                                 _Float16* __restrict__ out16) {
  __shared__ float s_sum[DM];
  __shared__ float s_sq[DM];
  int n = blockIdx.x;
  int c = threadIdx.x;
  float val = t32[n * DM + c];
  s_sum[c] = val;
  s_sq[c]  = val * val;
  __syncthreads();
  for (int off = DM / 2; off > 0; off >>= 1) {
    if (c < off) {
      s_sum[c] += s_sum[c + off];
      s_sq[c]  += s_sq[c + off];
    }
    __syncthreads();
  }
  float mu  = s_sum[0] * (1.0f / DM);
  float var = s_sq[0] * (1.0f / DM) - mu * mu;
  float r   = rsqrtf(var + 1e-5f);
  out16[n * DM + c] = (_Float16)((val - mu) * r * g[c] + b[c]);
}

// ---------------------------------------------------------------------------
// Epilogue for one 16x16 sub-tile (8 rows per lane-half).
// ---------------------------------------------------------------------------
__device__ __forceinline__ void epilogue_tile(
    v8f c, int mbase, int col, float bval, int Nt,
    const float* __restrict__ residual, float* __restrict__ outF,
    _Float16* __restrict__ outH, int act) {
#pragma unroll
  for (int r = 0; r < 8; ++r) {
    int m = mbase + r;
    float val = c[r] + bval;
    if (residual) val += residual[(size_t)m * Nt + col];
    if (act == 1) val = 0.5f * val * (1.0f + erff(val * 0.70710678118654752f));  // exact GELU
    if (outF) outF[(size_t)m * Nt + col] = val;
    if (outH) outH[(size_t)m * Nt + col] = (_Float16)val;
  }
}

// ---------------------------------------------------------------------------
// WMMA GEMM: C[M,Nt] = A16[M,K] x BT16[Nt,K]^T  (+bias +residual, act)
// One 32x32 tile per wave (2x2 register blocking): 4 independent
// v_wmma_f32_16x16x32_f16 per K-step, 2x arithmetic intensity vs 1-tile.
// ---------------------------------------------------------------------------
__global__ __launch_bounds__(128) void gemm_wmma_kernel(
    const _Float16* __restrict__ A, const _Float16* __restrict__ BT,
    const float* __restrict__ bias, const float* __restrict__ residual,
    float* __restrict__ outF, _Float16* __restrict__ outH,
    int M, int K, int Nt, int act) {
  int wave = blockIdx.x * (blockDim.x >> 5) + (threadIdx.x >> 5);
  int lane = threadIdx.x & 31;
  int tilesN = Nt >> 5;   // 32-wide tiles
  int tilesM = M >> 5;
  if (wave >= tilesM * tilesN) return;
  int tileM = wave / tilesN;
  int tileN = wave - tileM * tilesN;

  int row0  = tileM * 32 + (lane & 15);   // A fragment rows (M), second at +16
  int col0  = tileN * 32 + (lane & 15);   // B fragment cols (N), second at +16
  int khalf = (lane >> 4) * 8;            // per-lane K sub-offset (ISA 16-bit A layout)

  const _Float16* ap0 = A  + (size_t)row0 * K + khalf;
  const _Float16* ap1 = ap0 + (size_t)16 * K;
  const _Float16* bp0 = BT + (size_t)col0 * K + khalf;
  const _Float16* bp1 = bp0 + (size_t)16 * K;

  v8f c00 = {}, c01 = {}, c10 = {}, c11 = {};
  for (int kc = 0; kc < K; kc += 32) {
    FragU fa0, fa1, fb0, fb1;
    fa0.h[0] = *(const v8h*)(ap0 + kc);  fa0.h[1] = *(const v8h*)(ap0 + kc + 16);
    fa1.h[0] = *(const v8h*)(ap1 + kc);  fa1.h[1] = *(const v8h*)(ap1 + kc + 16);
    fb0.h[0] = *(const v8h*)(bp0 + kc);  fb0.h[1] = *(const v8h*)(bp0 + kc + 16);
    fb1.h[0] = *(const v8h*)(bp1 + kc);  fb1.h[1] = *(const v8h*)(bp1 + kc + 16);
    if (kc + 64 < K) {
      __builtin_prefetch((const void*)(ap0 + kc + 64), 0, 1);  // global_prefetch_b8
      __builtin_prefetch((const void*)(ap1 + kc + 64), 0, 1);
      __builtin_prefetch((const void*)(bp0 + kc + 64), 0, 1);
      __builtin_prefetch((const void*)(bp1 + kc + 64), 0, 1);
    }
    c00 = __builtin_amdgcn_wmma_f32_16x16x32_f16(false, fa0.f, false, fb0.f, (short)0, c00, false, false);
    c01 = __builtin_amdgcn_wmma_f32_16x16x32_f16(false, fa0.f, false, fb1.f, (short)0, c01, false, false);
    c10 = __builtin_amdgcn_wmma_f32_16x16x32_f16(false, fa1.f, false, fb0.f, (short)0, c10, false, false);
    c11 = __builtin_amdgcn_wmma_f32_16x16x32_f16(false, fa1.f, false, fb1.f, (short)0, c11, false, false);
  }

  // C layout: VGPR r -> M = r + 8*(lane>=16); lane&15 -> N
  int mb0 = tileM * 32 + (lane >> 4) * 8;
  int mb1 = mb0 + 16;
  float bv0 = bias ? bias[col0] : 0.0f;
  float bv1 = bias ? bias[col0 + 16] : 0.0f;
  epilogue_tile(c00, mb0, col0,      bv0, Nt, residual, outF, outH, act);
  epilogue_tile(c01, mb0, col0 + 16, bv1, Nt, residual, outF, outH, act);
  epilogue_tile(c10, mb1, col0,      bv0, Nt, residual, outF, outH, act);
  epilogue_tile(c11, mb1, col0 + 16, bv1, Nt, residual, outF, outH, act);
}

// ---------------------------------------------------------------------------
// Windowed attention: one thread per (token, head). 27 neighbors, hd=32.
// Neighbor indices + mask computed from geometry; padding -> masked & zero.
// ---------------------------------------------------------------------------
__global__ void attn_kernel(const _Float16* __restrict__ q16,
                            const _Float16* __restrict__ k16,
                            const _Float16* __restrict__ v16,
                            const int* __restrict__ known_mask,
                            _Float16* __restrict__ o16) {
  int gid = blockIdx.x * blockDim.x + threadIdx.x;
  if (gid >= NTOK * NH) return;
  int n    = gid >> 3;
  int head = gid & 7;
  int b    = n / VOX;
  int cell = n - b * VOX;
  int z = cell / 9, y = (cell / 3) % 3, x = cell % 3;
  int hoff = head * HD;

  float qv[HD];
  const _Float16* qp = q16 + (size_t)n * DM + hoff;
#pragma unroll
  for (int d = 0; d < HD; ++d) qv[d] = (float)qp[d];

  const float scale = 0.17677669529663687f;  // 1/sqrt(32)
  float s[VOX];
  float mk[VOX];
  int   idx[VOX];
#pragma unroll
  for (int l = 0; l < VOX; ++l) {
    int dz = l / 9 - 1, dy = (l / 3) % 3 - 1, dx = l % 3 - 1;
    int nz = z + dz, ny = y + dy, nx = x + dx;
    bool inb = (unsigned)nz < 3u && (unsigned)ny < 3u && (unsigned)nx < 3u;
    int nn = inb ? (b * VOX + nz * 9 + ny * 3 + nx) : 0;
    bool keep = inb && (known_mask[nn] != 0);
    float dot = 0.0f;
    if (keep) {
      const _Float16* kp = k16 + (size_t)nn * DM + hoff;
#pragma unroll
      for (int d = 0; d < HD; ++d) dot += qv[d] * (float)kp[d];
    }
    s[l]   = keep ? dot * scale : -1e30f;
    mk[l]  = keep ? 1.0f : 0.0f;
    idx[l] = nn;
  }

  float mx = -1e30f;
#pragma unroll
  for (int l = 0; l < VOX; ++l) mx = fmaxf(mx, s[l]);
  float sum = 0.0f;
  float a[VOX];
#pragma unroll
  for (int l = 0; l < VOX; ++l) { a[l] = expf(s[l] - mx); sum += a[l]; }
  float inv = 1.0f / sum;
#pragma unroll
  for (int l = 0; l < VOX; ++l) a[l] *= inv * mk[l];  // fully-masked row -> all zeros

  float od[HD];
#pragma unroll
  for (int d = 0; d < HD; ++d) od[d] = 0.0f;
#pragma unroll
  for (int l = 0; l < VOX; ++l) {
    if (mk[l] != 0.0f) {
      const _Float16* vp = v16 + (size_t)idx[l] * DM + hoff;
      float al = a[l];
#pragma unroll
      for (int d = 0; d < HD; ++d) od[d] += al * (float)vp[d];
    }
  }
  _Float16* op = o16 + (size_t)n * DM + hoff;
#pragma unroll
  for (int d = 0; d < HD; ++d) op[d] = (_Float16)od[d];
}

// ---------------------------------------------------------------------------
// Output head: out[b] = dot(t[center], w_out) + b_out. LDS reduction.
// ---------------------------------------------------------------------------
__global__ void head_kernel(const float* __restrict__ t32,
                            const float* __restrict__ w_out,
                            const float* __restrict__ b_out,
                            float* __restrict__ out) {
  __shared__ float red[DM];
  int b = blockIdx.x;
  int c = threadIdx.x;
  int n = b * VOX + 13;  // center voxel (1,1,1)
  red[c] = t32[n * DM + c] * w_out[c];
  __syncthreads();
  for (int off = DM / 2; off > 0; off >>= 1) {
    if (c < off) red[c] += red[c + off];
    __syncthreads();
  }
  if (c == 0) out[b] = red[0] + b_out[0];
}

// ---------------------------------------------------------------------------
extern "C" void kernel_launch(void* const* d_in, const int* in_sizes, int n_in,
                              void* d_out, int out_size, void* d_ws, size_t ws_size,
                              hipStream_t stream) {
  const float* patch   = (const float*)d_in[0];
  const int*   kmask   = (const int*)  d_in[1];
  const float* w_in    = (const float*)d_in[2];
  const float* b_in    = (const float*)d_in[3];
  const float* pos     = (const float*)d_in[4];
  const float* ln1_g   = (const float*)d_in[5];
  const float* ln1_b   = (const float*)d_in[6];
  const float* ln2_g   = (const float*)d_in[7];
  const float* ln2_b   = (const float*)d_in[8];
  const float* Wq      = (const float*)d_in[9];
  const float* bq      = (const float*)d_in[10];
  const float* Wk      = (const float*)d_in[11];
  const float* bk      = (const float*)d_in[12];
  const float* Wv      = (const float*)d_in[13];
  const float* bv      = (const float*)d_in[14];
  const float* Wo      = (const float*)d_in[15];
  const float* bo      = (const float*)d_in[16];
  const float* W1      = (const float*)d_in[17];
  const float* b1      = (const float*)d_in[18];
  const float* W2      = (const float*)d_in[19];
  const float* b2      = (const float*)d_in[20];
  const float* w_out   = (const float*)d_in[21];
  const float* b_out   = (const float*)d_in[22];
  float* out = (float*)d_out;

  // ---- workspace carve-up (256B aligned) ----
  char* base = (char*)d_ws;
  size_t off = 0;
  auto carve = [&](size_t bytes) -> char* {
    char* p = base + off;
    off = (off + bytes + 255) & ~(size_t)255;
    return p;
  };
  const size_t WPL = (size_t)4 * DM * DM + (size_t)DM * FF + (size_t)FF * DM;  // halfs/layer
  _Float16* wts  = (_Float16*)carve(NLAYERS * WPL * sizeof(_Float16));
  float*    t32  = (float*)   carve((size_t)NTOK * DM * sizeof(float));
  _Float16* t16  = (_Float16*)carve((size_t)NTOK * DM * sizeof(_Float16));
  _Float16* tn16 = (_Float16*)carve((size_t)NTOK * DM * sizeof(_Float16));  // also reused as o16
  _Float16* q16  = (_Float16*)carve((size_t)NTOK * DM * sizeof(_Float16));
  _Float16* k16  = (_Float16*)carve((size_t)NTOK * DM * sizeof(_Float16));
  _Float16* v16  = (_Float16*)carve((size_t)NTOK * DM * sizeof(_Float16));
  _Float16* h16  = (_Float16*)carve((size_t)NTOK * FF * sizeof(_Float16));
  (void)ws_size; (void)n_in; (void)in_sizes; (void)out_size;

  auto pack = [&](const float* src, _Float16* dst, int K, int Nt) {
    int cnt = K * Nt;
    pack_wT_kernel<<<(cnt + 255) / 256, 256, 0, stream>>>(src, dst, K, Nt);
  };
  auto gemm = [&](const _Float16* A, const _Float16* BT, const float* bias,
                  const float* res, float* oF, _Float16* oH,
                  int M, int K, int Nt, int act) {
    int tiles = (M / 32) * (Nt / 32);
    gemm_wmma_kernel<<<(tiles + 3) / 4, 128, 0, stream>>>(A, BT, bias, res, oF, oH, M, K, Nt, act);
  };

  // ---- pack all weights (transposed f16) ----
  for (int l = 0; l < NLAYERS; ++l) {
    _Float16* wl = wts + l * WPL;
    pack(Wq + (size_t)l * DM * DM, wl + 0 * DM * DM, DM, DM);
    pack(Wk + (size_t)l * DM * DM, wl + 1 * DM * DM, DM, DM);
    pack(Wv + (size_t)l * DM * DM, wl + 2 * DM * DM, DM, DM);
    pack(Wo + (size_t)l * DM * DM, wl + 3 * DM * DM, DM, DM);
    pack(W1 + (size_t)l * DM * FF, wl + 4 * DM * DM, DM, FF);
    pack(W2 + (size_t)l * FF * DM, wl + 4 * DM * DM + (size_t)DM * FF, FF, DM);
  }

  // ---- embed ----
  embed_kernel<<<NTOK, DM, 0, stream>>>(patch, w_in, b_in, pos, t32, t16);

  // ---- layers ----
  for (int l = 0; l < NLAYERS; ++l) {
    _Float16* WqT = wts + l * WPL + 0 * DM * DM;
    _Float16* WkT = wts + l * WPL + 1 * DM * DM;
    _Float16* WvT = wts + l * WPL + 2 * DM * DM;
    _Float16* WoT = wts + l * WPL + 3 * DM * DM;
    _Float16* W1T = wts + l * WPL + 4 * DM * DM;
    _Float16* W2T = wts + l * WPL + 4 * DM * DM + (size_t)DM * FF;

    // LN1 -> tn16 ; Q from normed, K/V from raw t (linearity of gather)
    layernorm_kernel<<<NTOK, DM, 0, stream>>>(t32, ln1_g + l * DM, ln1_b + l * DM, tn16);
    gemm(tn16, WqT, bq + l * DM, nullptr, nullptr, q16, NTOK, DM, DM, 0);
    gemm(t16,  WkT, bk + l * DM, nullptr, nullptr, k16, NTOK, DM, DM, 0);
    gemm(t16,  WvT, bv + l * DM, nullptr, nullptr, v16, NTOK, DM, DM, 0);

    // attention -> o16 (reuse tn16 buffer; q GEMM already consumed it)
    attn_kernel<<<(NTOK * NH + 255) / 256, 256, 0, stream>>>(q16, k16, v16, kmask, tn16);

    // Wo with fused residual: t = t + o@Wo + bo  (writes f32 + f16 mirrors)
    gemm(tn16, WoT, bo + l * DM, t32, t32, t16, NTOK, DM, DM, 0);

    // LN2 -> tn16 ; FFN1 (GELU) -> h16 ; FFN2 with residual -> t
    layernorm_kernel<<<NTOK, DM, 0, stream>>>(t32, ln2_g + l * DM, ln2_b + l * DM, tn16);
    gemm(tn16, W1T, b1 + l * FF, nullptr, nullptr, h16, NTOK, DM, FF, 1);
    gemm(h16,  W2T, b2 + l * DM, t32, t32, t16, NTOK, FF, DM, 0);
  }

  // ---- output head ----
  head_kernel<<<NB, DM, 0, stream>>>(t32, w_out, b_out, out);
}
